// STGCN_31894427140601
// MI455X (gfx1250) — matrix operand order repaired and checked by reading, not compiled, not run on hardware
//
#include <hip/hip_runtime.h>
#include <hip/hip_bf16.h>
#include <math.h>

// ---------------------------------------------------------------------------
// Types for CDNA5 WMMA (wave32): A/B = 16 bf16 per lane (8 VGPRs),
// C/D = 8 f32 per lane.
// ---------------------------------------------------------------------------
typedef __attribute__((ext_vector_type(16))) __bf16 v16bf;
typedef __attribute__((ext_vector_type(8)))  float  v8f;

#define IN_C 128
#define HID  128

__device__ __forceinline__ __bf16 f2bf(float f) {
    union { float f; unsigned u; } in; in.f = f;
    unsigned u = in.u;
    unsigned r = u + 0x7FFFu + ((u >> 16) & 1u);   // round-to-nearest-even
    union { unsigned short s; __bf16 b; } out;
    out.s = (unsigned short)(r >> 16);
    return out.b;
}

// ---------------------------------------------------------------------------
// Small utility kernels
// ---------------------------------------------------------------------------
__global__ void k_fill1(float* p, int n) {
    int i = blockIdx.x * blockDim.x + threadIdx.x;
    if (i < n) p[i] = 1.0f;                         // self-loop contribution
}

__global__ void k_deg(const long long* __restrict__ dst, float* __restrict__ deg, int E) {
    int e = blockIdx.x * blockDim.x + threadIdx.x;
    if (e < E) atomicAdd(&deg[(int)dst[e]], 1.0f);
}

__global__ void k_rsqrt(const float* __restrict__ deg, float* __restrict__ dinv, int n) {
    int i = blockIdx.x * blockDim.x + threadIdx.x;
    if (i < n) dinv[i] = rsqrtf(deg[i]);
}

__global__ void k_bsum(const float* __restrict__ a, const float* __restrict__ b,
                       float* __restrict__ o, int n) {
    int i = blockIdx.x * blockDim.x + threadIdx.x;
    if (i < n) o[i] = a[i] + b[i];
}

// ---------------------------------------------------------------------------
// Pack an [Nrows x 128] f32 matrix into per-lane WMMA A-fragments (bf16),
// optionally fused with +bias and ReLU (used for the post-scatter repack).
// Fragment layout (16-bit A 16x32, wave32):
//   lane<16 : row = rt*16+lane,    K = kt*32 + {0..7, 16..23}
//   lane>=16: row = rt*16+lane-16, K = kt*32 + {8..15, 24..31}
// dst index = ((rt*4 + kt)*32 + lane), 32 contiguous bytes per lane.
// ---------------------------------------------------------------------------
__global__ void k_pack_rows(const float* __restrict__ src, const float* __restrict__ bias,
                            int relu, v16bf* __restrict__ dst, int total) {
    int tid = blockIdx.x * blockDim.x + threadIdx.x;
    if (tid >= total) return;
    int lane = tid & 31;
    int kt   = (tid >> 5) & 3;
    int rt   = tid >> 7;
    int row  = rt * 16 + (lane & 15);
    int col0 = kt * 32 + (lane >> 4) * 8;
    const float* r = src + (size_t)row * 128;
    v16bf frag;
#pragma unroll
    for (int e = 0; e < 16; ++e) {
        int c = (e < 8) ? (col0 + e) : (col0 + 8 + e);   // {0..7} then {16..23}
        float v = r[c];
        if (bias) v += bias[c];
        if (relu) v = fmaxf(v, 0.f);
        frag[e] = f2bf(v);
    }
    dst[tid] = frag;
}

// ---------------------------------------------------------------------------
// Pack a weight matrix into per-lane WMMA B-fragments (bf16).
// Logical B is [K=128 x Nn]; element B[k][n] = trans ? W[n*128+k] : W[k*Nn+n].
// Fragment layout (16-bit B 32x16, wave32):
//   lane<16 : N = nt*16+lane,    K = kt*32 + {0..7, 16..23}
//   lane>=16: N = nt*16+lane-16, K = kt*32 + {8..15, 24..31}
// dst index = ((nt*4 + kt)*32 + lane)
// ---------------------------------------------------------------------------
__global__ void k_pack_w(const float* __restrict__ W, v16bf* __restrict__ dst,
                         int Nn, int trans, int total) {
    int tid = blockIdx.x * blockDim.x + threadIdx.x;
    if (tid >= total) return;
    int lane  = tid & 31;
    int kt    = (tid >> 5) & 3;
    int nt    = tid >> 7;
    int n     = nt * 16 + (lane & 15);
    int kbase = kt * 32 + (lane >> 4) * 8;
    v16bf frag;
#pragma unroll
    for (int e = 0; e < 16; ++e) {
        int K = kbase + e + ((e < 8) ? 0 : 8);
        float v = trans ? W[(size_t)n * 128 + K] : W[(size_t)K * Nn + n];
        frag[e] = f2bf(v);
    }
    dst[tid] = frag;
}

// ---------------------------------------------------------------------------
// WMMA GEMM: C[M x Nn] = A[M x 128] * B[128 x Nn] (+bias, optional ReLU)
// One wave computes a 16x64 strip: 4 n-tiles x 4 k-steps = 16 chained
// v_wmma_f32_16x16x32_bf16, with the four A fragments held in VGPRs and
// reused across all 4 n-tiles (4x less A traffic from HBM/L2).
// block = 256 threads (8 waves -> 8 consecutive row tiles), grid.y = Nn/64.
// ---------------------------------------------------------------------------
__global__ void k_gemm(const v16bf* __restrict__ Apack, const v16bf* __restrict__ Bpack,
                       float* __restrict__ C, int RT, int Nn,
                       const float* __restrict__ bias, int relu) {
    int wave = threadIdx.x >> 5;
    int lane = threadIdx.x & 31;
    int rt   = blockIdx.x * 8 + wave;
    if (rt >= RT) return;                  // uniform per-wave exit: EXEC all-ones for WMMA
    int ng   = blockIdx.y;                 // group of 4 consecutive n-tiles

    const v16bf* Af = Apack + (size_t)rt * 4 * 32;
    v16bf a[4];
#pragma unroll
    for (int kt = 0; kt < 4; ++kt) a[kt] = Af[kt * 32 + lane];

    v8f acc[4] = {};
#pragma unroll
    for (int s = 0; s < 4; ++s) {
        const v16bf* Bf = Bpack + (size_t)(ng * 4 + s) * 4 * 32;
#pragma unroll
        for (int kt = 0; kt < 4; ++kt) {
            v16bf b = Bf[kt * 32 + lane];
            acc[s] = __builtin_amdgcn_wmma_f32_16x16x32_bf16(
                /*neg_a=*/false, a[kt], /*neg_b=*/false, b,
                /*c_mod=*/(short)0, acc[s], /*reuse_a=*/false, /*reuse_b=*/false);
        }
    }

    // D layout: lane l, VGPR v -> M = v + 8*(l>=16), N = l%16
    int hi = lane >> 4;
    int m0 = rt * 16 + 8 * hi;
#pragma unroll
    for (int s = 0; s < 4; ++s) {
        int n  = (ng * 4 + s) * 16 + (lane & 15);
        float bv = bias ? bias[n] : 0.f;
        union { v8f v; float f[8]; } r; r.v = acc[s];
#pragma unroll
        for (int v = 0; v < 8; ++v) {
            float val = r.f[v] + bv;
            if (relu) val = fmaxf(val, 0.f);
            C[(size_t)(m0 + v) * Nn + n] = val;
        }
    }
}

// ---------------------------------------------------------------------------
// GCN self-loop init: seg[i,:] = hpre[i,:] * dinv[i]^2
// ---------------------------------------------------------------------------
__global__ void k_self(const float* __restrict__ hpre, const float* __restrict__ dinv,
                       float* __restrict__ seg, int total) {
    int t = blockIdx.x * blockDim.x + threadIdx.x;
    if (t >= total) return;
    int i = t >> 7;                         // / 128
    float w = dinv[i]; w *= w;
    seg[t] = hpre[t] * w;
}

// ---------------------------------------------------------------------------
// Edge scatter: seg[dst,:] += hpre[src,:] * dinv[src]*dinv[dst]
// One thread per (edge, 4-float chunk): 32 threads per edge, float4 loads,
// f32 atomics resolved at L2.
// ---------------------------------------------------------------------------
__global__ void k_scatter(const long long* __restrict__ src, const long long* __restrict__ dst,
                          const float* __restrict__ dinv, const float* __restrict__ hpre,
                          float* __restrict__ seg, int E) {
    long long t = (long long)blockIdx.x * blockDim.x + threadIdx.x;
    long long e = t >> 5;
    if (e >= E) return;
    int q = (int)(t & 31);
    int s = (int)src[e];
    int d = (int)dst[e];
    float w = dinv[s] * dinv[d];
    float4 v = reinterpret_cast<const float4*>(hpre)[(size_t)s * 32 + q];
    float* o = seg + (size_t)d * 128 + q * 4;
    atomicAdd(o + 0, v.x * w);
    atomicAdd(o + 1, v.y * w);
    atomicAdd(o + 2, v.z * w);
    atomicAdd(o + 3, v.w * w);
}

// ---------------------------------------------------------------------------
// Persistent LSTM scan. One workgroup, 512 threads (16 wave32).
// W_hh (512x128 f32 = 256KB) lives in LDS -- only possible on CDNA5's
// 320KB-per-WGP LDS -- in a float4-grouped layout:
//   wt4[g*512 + j] = Whh[j][4g..4g+3]   (g = 0..31)
// so the 128-MAC dot product is 32 x (ds_load_b128 + broadcast ds_load_b128
// + 4 FMA). Thread j owns gate element j; threads 0..127 own (c,h) lanes.
// xp is register-prefetched 1 step ahead and cache-prefetched 2 ahead.
// ---------------------------------------------------------------------------
__global__ void k_lstm(const float* __restrict__ xp, const float* __restrict__ Whh,
                       float* __restrict__ hout, int T) {
    extern __shared__ float sm[];
    float4* wt4 = (float4*)sm;        // [32][512] float4
    float*  hs  = sm + 128 * 512;     // [128] current hidden state
    float*  gs  = hs + 128;           // [512] activated gates
    int j = threadIdx.x;

    const float4* W4 = (const float4*)Whh;
    for (int g = 0; g < 32; ++g) wt4[g * 512 + j] = W4[(size_t)j * 32 + g];
    if (j < 128) hs[j] = 0.f;
    float c = 0.f;
    __syncthreads();

    const float4* hs4 = (const float4*)hs;
    float xpn = xp[j];
    for (int t = 0; t < T; ++t) {
        float acc = xpn;
        if (t + 1 < T) xpn = xp[(size_t)(t + 1) * 512 + j];          // register prefetch
        if (t + 2 < T) __builtin_prefetch(&xp[(size_t)(t + 2) * 512 + j], 0, 0);
#pragma unroll 8
        for (int g = 0; g < 32; ++g) {
            float4 wv = wt4[g * 512 + j];
            float4 hv = hs4[g];                                       // LDS broadcast
            acc = fmaf(hv.x, wv.x, acc);
            acc = fmaf(hv.y, wv.y, acc);
            acc = fmaf(hv.z, wv.z, acc);
            acc = fmaf(hv.w, wv.w, acc);
        }
        // gate order i,f,g,o : g (j in [256,384)) uses tanh, others sigmoid
        float act = (j >= 256 && j < 384) ? tanhf(acc)
                                          : (1.f / (1.f + __expf(-acc)));
        gs[j] = act;
        __syncthreads();
        if (j < 128) {
            float ig = gs[j], fg = gs[128 + j], gg = gs[256 + j], og = gs[384 + j];
            c = fg * c + ig * gg;
            float h = og * tanhf(c);
            hs[j] = h;
            hout[(size_t)t * 128 + j] = h;
        }
        __syncthreads();
    }
}

// ---------------------------------------------------------------------------
// Host-side orchestration
// ---------------------------------------------------------------------------
static inline int ceil_div(long long a, long long b) { return (int)((a + b - 1) / b); }

extern "C" void kernel_launch(void* const* d_in, const int* in_sizes, int n_in,
                              void* d_out, int out_size, void* d_ws, size_t ws_size,
                              hipStream_t stream) {
    const float*     x    = (const float*)d_in[0];
    const long long* ei   = (const long long*)d_in[1];   // int64 [2, E]
    const float*     W1   = (const float*)d_in[3];
    const float*     b1   = (const float*)d_in[4];
    const float*     W2   = (const float*)d_in[5];
    const float*     b2   = (const float*)d_in[6];
    const float*     Wih  = (const float*)d_in[7];       // [512,128]
    const float*     Whh  = (const float*)d_in[8];       // [512,128]
    const float*     bih  = (const float*)d_in[9];
    const float*     bhh  = (const float*)d_in[10];
    const float*     Wlin = (const float*)d_in[11];      // [64,128]
    const float*     blin = (const float*)d_in[12];
    float*           out  = (float*)d_out;

    const int N  = in_sizes[0] / IN_C;      // 50000
    const int E  = in_sizes[1] / 2;         // 1600000
    const int RT = N / 16;                  // 3125 row tiles (exact)
    const long long* esrc = ei;
    const long long* edst = ei + E;

    // ---- workspace layout ----
    char*  base = (char*)d_ws;
    size_t off  = 0;
    auto give = [&](size_t bytes) -> void* {
        void* p = base + off;
        off += (bytes + 255) & ~(size_t)255;
        return p;
    };
    float* deg      = (float*)give((size_t)N * 4);
    float* dinv     = (float*)give((size_t)N * 4);
    float* bsum     = (float*)give(512 * 4);
    v16bf* packA    = (v16bf*)give((size_t)RT * 4 * 32 * sizeof(v16bf));   // 12.8 MB
    v16bf* packB1   = (v16bf*)give((size_t)(128/16) * 4 * 32 * sizeof(v16bf));
    v16bf* packB2   = (v16bf*)give((size_t)(128/16) * 4 * 32 * sizeof(v16bf));
    v16bf* packBih  = (v16bf*)give((size_t)(512/16) * 4 * 32 * sizeof(v16bf));
    v16bf* packBlin = (v16bf*)give((size_t)(64/16)  * 4 * 32 * sizeof(v16bf));
    float* hpre     = (float*)give((size_t)N * 128 * 4);                   // 25.6 MB
    float* seg      = (float*)give((size_t)N * 128 * 4);                   // 25.6 MB
    float* xpbuf    = (float*)give((size_t)N * 512 * 4);                   // 102.4 MB
    float* hl       = (float*)give((size_t)N * 128 * 4);                   // 25.6 MB
    (void)ws_size; (void)n_in;

    const int BLK = 256;

    // ---- GCN normalization ----
    k_fill1<<<ceil_div(N, BLK), BLK, 0, stream>>>(deg, N);
    k_deg  <<<ceil_div(E, BLK), BLK, 0, stream>>>(edst, deg, E);
    k_rsqrt<<<ceil_div(N, BLK), BLK, 0, stream>>>(deg, dinv, N);
    k_bsum <<<2, BLK, 0, stream>>>(bih, bhh, bsum, 512);

    // ---- weight packing (bf16 WMMA B-fragments) ----
    k_pack_w<<<ceil_div((128/16)*128, BLK), BLK, 0, stream>>>(W1,   packB1,  128, 0, (128/16)*128);
    k_pack_w<<<ceil_div((128/16)*128, BLK), BLK, 0, stream>>>(W2,   packB2,  128, 0, (128/16)*128);
    k_pack_w<<<ceil_div((512/16)*128, BLK), BLK, 0, stream>>>(Wih,  packBih, 512, 1, (512/16)*128);
    k_pack_w<<<ceil_div((64/16)*128,  BLK), BLK, 0, stream>>>(Wlin, packBlin, 64, 1, (64/16)*128);

    const int packTot = RT * 128;
    const long long scatTot = (long long)E * 32;
    dim3 g128(ceil_div(RT, 8), 128 / 64);   // 2 n-groups of 64 cols
    dim3 g512(ceil_div(RT, 8), 512 / 64);   // 8 n-groups
    dim3 g64 (ceil_div(RT, 8),  64 / 64);   // 1 n-group

    // ---- GCN layer 1 ----
    k_pack_rows<<<ceil_div(packTot, BLK), BLK, 0, stream>>>(x, nullptr, 0, packA, packTot);
    k_gemm<<<g128, BLK, 0, stream>>>(packA, packB1, hpre, RT, 128, nullptr, 0);
    k_self<<<ceil_div((long long)N * 128, BLK), BLK, 0, stream>>>(hpre, dinv, seg, N * 128);
    k_scatter<<<ceil_div(scatTot, BLK), BLK, 0, stream>>>(esrc, edst, dinv, hpre, seg, E);
    k_pack_rows<<<ceil_div(packTot, BLK), BLK, 0, stream>>>(seg, b1, 1, packA, packTot); // +b1, ReLU

    // ---- GCN layer 2 ----
    k_gemm<<<g128, BLK, 0, stream>>>(packA, packB2, hpre, RT, 128, nullptr, 0);
    k_self<<<ceil_div((long long)N * 128, BLK), BLK, 0, stream>>>(hpre, dinv, seg, N * 128);
    k_scatter<<<ceil_div(scatTot, BLK), BLK, 0, stream>>>(esrc, edst, dinv, hpre, seg, E);
    k_pack_rows<<<ceil_div(packTot, BLK), BLK, 0, stream>>>(seg, b2, 1, packA, packTot); // +b2, ReLU

    // ---- LSTM input projection: xp = h2 @ W_ih^T + (b_ih + b_hh) ----
    k_gemm<<<g512, BLK, 0, stream>>>(packA, packBih, xpbuf, RT, 512, bsum, 0);

    // ---- sequential LSTM scan (W_hh resident in 320KB WGP LDS) ----
    const size_t lstm_smem = (size_t)(128 * 512 + 128 + 512) * sizeof(float); // ~258.5 KB
    k_lstm<<<1, 512, lstm_smem, stream>>>(xpbuf, Whh, hl, N);

    // ---- output head: out = h @ W_lin^T + b_lin ----
    k_pack_rows<<<ceil_div(packTot, BLK), BLK, 0, stream>>>(hl, nullptr, 0, packA, packTot);
    k_gemm<<<g64, BLK, 0, stream>>>(packA, packBlin, out, RT, 64, blin, 0);

    (void)out_size;
}
